// GatedAttention_19129784337095
// MI455X (gfx1250) — compile-verified, matrix operand order checked by reading, not run on hardware
//
#include <hip/hip_runtime.h>
#include <hip/hip_bf16.h>
#include <math.h>

typedef __bf16 bf16_t;
typedef __attribute__((ext_vector_type(8)))  __bf16 v8bf;
typedef __attribute__((ext_vector_type(16))) __bf16 v16bf;
typedef __attribute__((ext_vector_type(8)))  float  v8f;

#define N_HEADS   16
#define HEAD_DIM  128
#define EMBD      2048
#define KV_HEADS  4
#define SEQ_T     2048
#define BATCH_B   2

// ---------- helpers ----------

__device__ __forceinline__ v8f zero8() {
  v8f z;
#pragma unroll
  for (int i = 0; i < 8; ++i) z[i] = 0.0f;
  return z;
}

// Build a 16x16x32 bf16 A/B fragment from a row-contiguous bf16 buffer.
// CDNA5 16-bit fragment layout: element e -> K = (e<8?0:16) + half*8 + (e%8),
// so elements 0..7 and 8..15 are each 8 contiguous bf16 (16B) loads.
__device__ __forceinline__ v16bf make_frag(const bf16_t* rowptr, int half) {
  v8bf lo = *(const v8bf*)(rowptr + half * 8);
  v8bf hi = *(const v8bf*)(rowptr + 16 + half * 8);
  return __builtin_shufflevector(lo, hi,
      0, 1, 2, 3, 4, 5, 6, 7, 8, 9, 10, 11, 12, 13, 14, 15);
}

// ---------- fp32 -> bf16 convert ----------

__global__ void ga_cvt_bf16(const float* __restrict__ src,
                            bf16_t* __restrict__ dst, int n) {
  for (int i = blockIdx.x * blockDim.x + threadIdx.x; i < n;
       i += gridDim.x * blockDim.x)
    dst[i] = (bf16_t)src[i];
}

// ---------- generic WMMA GEMM: C[M,N] = A[M,K] * B[N,K]^T ----------
// block = 128 threads (4 waves); tile 64x64; K-step 32.

__global__ __launch_bounds__(128) void ga_gemm_wmma(
    const bf16_t* __restrict__ A, const bf16_t* __restrict__ Bw,
    float* __restrict__ C, int M, int N, int K) {
  __shared__ alignas(16) bf16_t As[64][32];
  __shared__ alignas(16) bf16_t Bs[64][32];

  const int tid = threadIdx.x;
  const int wave = tid >> 5;
  const int lane = tid & 31;
  const int nL = lane & 15;
  const int half = lane >> 4;
  const int m0 = blockIdx.x * 64;
  const int n0 = blockIdx.y * 64;

  v8f acc[4];
#pragma unroll
  for (int i = 0; i < 4; ++i) acc[i] = zero8();

  const int ksteps = K >> 5;
  for (int kt = 0; kt < ksteps; ++kt) {
    // cooperative loads: each tile 64x32 bf16 = 256 groups of 8 elems
#pragma unroll
    for (int g = 0; g < 2; ++g) {
      int idx = tid + g * 128;        // 0..255
      int row = idx >> 2;
      int col = (idx & 3) * 8;
      *(v8bf*)&As[row][col] =
          *(const v8bf*)&A[(size_t)(m0 + row) * K + kt * 32 + col];
      *(v8bf*)&Bs[row][col] =
          *(const v8bf*)&Bw[(size_t)(n0 + row) * K + kt * 32 + col];
    }
    __syncthreads();

    v16bf af = make_frag(&As[wave * 16 + nL][0], half);
#pragma unroll
    for (int nt = 0; nt < 4; ++nt) {
      v16bf bfr = make_frag(&Bs[nt * 16 + nL][0], half);
      acc[nt] = __builtin_amdgcn_wmma_f32_16x16x32_bf16(
          false, af, false, bfr, (short)0, acc[nt], false, false);
    }
    __syncthreads();
  }

#pragma unroll
  for (int nt = 0; nt < 4; ++nt)
#pragma unroll
    for (int r = 0; r < 8; ++r) {
      int m = m0 + wave * 16 + half * 8 + r;
      int n = n0 + nt * 16 + nL;
      C[(size_t)m * N + n] = acc[nt][r];
    }
}

// ---------- per-head RMSNorm + RoPE, repack to [B, H, T, hs] bf16 ----------
// blockIdx.y: 0..15 -> q head, 16..19 -> k head, 20..23 -> v head

__global__ __launch_bounds__(128) void ga_norm_rope(
    const float* __restrict__ qg, const float* __restrict__ kf,
    const float* __restrict__ vf, const float* __restrict__ qnw,
    const float* __restrict__ knw, bf16_t* __restrict__ qb,
    bf16_t* __restrict__ kb, bf16_t* __restrict__ vb) {
  __shared__ float sred[128];
  __shared__ float yv[128];
  const int m = blockIdx.x;   // b*T + t
  const int hh = blockIdx.y;
  const int d = threadIdx.x;
  const int b = m / SEQ_T;
  const int t = m % SEQ_T;

  const float* src;
  const float* w = nullptr;
  bf16_t* dst;
  bool dorope;
  if (hh < N_HEADS) {                      // q (first 128 of each 256 block)
    src = qg + (size_t)m * 4096 + hh * 256;
    w = qnw;
    dorope = true;
    dst = qb + (((size_t)b * N_HEADS + hh) * SEQ_T + t) * HEAD_DIM;
  } else if (hh < N_HEADS + KV_HEADS) {    // k
    int hk = hh - N_HEADS;
    src = kf + (size_t)m * 512 + hk * 128;
    w = knw;
    dorope = true;
    dst = kb + (((size_t)b * KV_HEADS + hk) * SEQ_T + t) * HEAD_DIM;
  } else {                                 // v: plain convert
    int hk = hh - N_HEADS - KV_HEADS;
    src = vf + (size_t)m * 512 + hk * 128;
    dorope = false;
    dst = vb + (((size_t)b * KV_HEADS + hk) * SEQ_T + t) * HEAD_DIM;
  }

  float x = src[d];
  if (!dorope) {          // block-uniform branch, safe w.r.t. barriers
    dst[d] = (bf16_t)x;
    return;
  }

  sred[d] = x * x;
  __syncthreads();
  for (int s = 64; s > 0; s >>= 1) {
    if (d < s) sred[d] += sred[d + s];
    __syncthreads();
  }
  float rms = rsqrtf(sred[0] * (1.0f / 128.0f) + 1e-6f);
  float y = x * rms * (1.0f + w[d]);
  yv[d] = y;
  __syncthreads();

  int f = d & 63;
  float invf = powf(10000.0f, -(float)f * (1.0f / 64.0f));
  float ang = (float)t * invf;
  float c = cosf(ang), s = sinf(ang);
  float rot = (d < 64) ? -yv[d + 64] : yv[d - 64];
  dst[d] = (bf16_t)(y * c + rot * s);
}

// ---------- flash attention: causal GQA, WMMA QK^T and PV ----------
// grid (T/64, H, B), 128 threads = 4 waves; wave handles 16 query rows.

__global__ __launch_bounds__(128) void ga_flash_attn(
    const bf16_t* __restrict__ qb, const bf16_t* __restrict__ kb,
    const bf16_t* __restrict__ vb, float* __restrict__ attn) {
  __shared__ alignas(16) bf16_t Kt[64][128];    // key tile, row-major
  __shared__ alignas(16) bf16_t VtT[128][64];   // value tile, transposed
  __shared__ alignas(16) bf16_t Pl[4][16][64];  // per-wave P staging

  const int tid = threadIdx.x;
  const int wave = tid >> 5;
  const int lane = tid & 31;
  const int nL = lane & 15;
  const int half = lane >> 4;
  const int q0 = blockIdx.x * 64;
  const int h = blockIdx.y;
  const int b = blockIdx.z;
  const int hk = h >> 2;   // GQA: 4 query heads per kv head

  const bf16_t* qbase = qb + ((size_t)b * N_HEADS + h) * SEQ_T * HEAD_DIM;
  const bf16_t* kbase = kb + ((size_t)b * KV_HEADS + hk) * SEQ_T * HEAD_DIM;
  const bf16_t* vbase = vb + ((size_t)b * KV_HEADS + hk) * SEQ_T * HEAD_DIM;

  // Q fragments for this wave's 16 rows, all 128 K-dim (4 chunks of 32)
  v16bf aq[4];
  {
    const bf16_t* qrow = qbase + (size_t)(q0 + wave * 16 + nL) * HEAD_DIM;
#pragma unroll
    for (int kc = 0; kc < 4; ++kc) aq[kc] = make_frag(qrow + kc * 32, half);
  }

  v8f o[8];
  float mrun[8], lrun[8];
#pragma unroll
  for (int i = 0; i < 8; ++i) {
    o[i] = zero8();
    mrun[i] = -1e30f;
    lrun[i] = 0.0f;
  }

  const float scale = 0.08838834764831845f;   // 1/sqrt(128)
  const int jend = q0 + 64;                   // causal cutoff (uniform per block)

  for (int j0 = 0; j0 < jend; j0 += 64) {
    // load K (row-major) and V (transposed) tiles: 64x128 bf16 each
#pragma unroll
    for (int g = 0; g < 8; ++g) {
      int idx = tid + g * 128;     // 0..1023
      int row = idx >> 4;
      int col = (idx & 15) * 8;
      *(v8bf*)&Kt[row][col] =
          *(const v8bf*)&kbase[(size_t)(j0 + row) * HEAD_DIM + col];
      v8bf vv = *(const v8bf*)&vbase[(size_t)(j0 + row) * HEAD_DIM + col];
#pragma unroll
      for (int e = 0; e < 8; ++e) VtT[col + e][row] = vv[e];
    }
    __syncthreads();

    // S = Q * K^T  (16 x 64 per wave)
    v8f st[4];
#pragma unroll
    for (int nt = 0; nt < 4; ++nt) {
      st[nt] = zero8();
#pragma unroll
      for (int kc = 0; kc < 4; ++kc) {
        v16bf bk = make_frag(&Kt[nt * 16 + nL][kc * 32], half);
        st[nt] = __builtin_amdgcn_wmma_f32_16x16x32_bf16(
            false, aq[kc], false, bk, (short)0, st[nt], false, false);
      }
    }

    // online softmax per row (rows live across 16 lanes of one half)
#pragma unroll
    for (int r = 0; r < 8; ++r) {
      int i = q0 + wave * 16 + half * 8 + r;
      float sv[4];
#pragma unroll
      for (int nt = 0; nt < 4; ++nt) {
        int j = j0 + nt * 16 + nL;
        float v = st[nt][r] * scale;
        sv[nt] = (j <= i) ? v : -1e30f;
      }
      float rmax = fmaxf(fmaxf(sv[0], sv[1]), fmaxf(sv[2], sv[3]));
#pragma unroll
      for (int msk = 1; msk <= 8; msk <<= 1)
        rmax = fmaxf(rmax, __shfl_xor(rmax, msk, 32));
      float mnew = fmaxf(mrun[r], rmax);
      float fac = __expf(mrun[r] - mnew);
      float psum = 0.0f;
#pragma unroll
      for (int nt = 0; nt < 4; ++nt) {
        float p = __expf(sv[nt] - mnew);
        psum += p;
        Pl[wave][half * 8 + r][nt * 16 + nL] = (bf16_t)p;
      }
#pragma unroll
      for (int msk = 1; msk <= 8; msk <<= 1)
        psum += __shfl_xor(psum, msk, 32);
      lrun[r] = lrun[r] * fac + psum;
      mrun[r] = mnew;
#pragma unroll
      for (int ot = 0; ot < 8; ++ot) o[ot][r] *= fac;
    }
    __syncthreads();

    // O += P * V   (16x64 times 64x128)
#pragma unroll
    for (int kc = 0; kc < 2; ++kc) {
      v16bf ap = make_frag(&Pl[wave][nL][kc * 32], half);
#pragma unroll
      for (int ot = 0; ot < 8; ++ot) {
        v16bf bv = make_frag(&VtT[ot * 16 + nL][kc * 32], half);
        o[ot] = __builtin_amdgcn_wmma_f32_16x16x32_bf16(
            false, ap, false, bv, (short)0, o[ot], false, false);
      }
    }
    __syncthreads();
  }

  // epilogue: normalize and write [B,T,H,hs] (== [B,T,C]) f32
#pragma unroll
  for (int r = 0; r < 8; ++r) {
    int i = q0 + wave * 16 + half * 8 + r;
    float invl = 1.0f / lrun[r];
    float* orow = attn + ((size_t)b * SEQ_T + i) * EMBD + h * HEAD_DIM;
#pragma unroll
    for (int ot = 0; ot < 8; ++ot) orow[ot * 16 + nL] = o[ot][r] * invl;
  }
}

// ---------- sigmoid gate + bf16 convert ----------

__global__ void ga_gate(const float* __restrict__ attn,
                        const float* __restrict__ qg,
                        bf16_t* __restrict__ gated, int total) {
  for (int idx = blockIdx.x * blockDim.x + threadIdx.x; idx < total;
       idx += gridDim.x * blockDim.x) {
    int m = idx >> 11;      // / 2048
    int c = idx & 2047;
    int hh = c >> 7;
    int d = c & 127;
    float g = qg[(size_t)m * 4096 + hh * 256 + 128 + d];  // gate half
    float a = attn[idx];
    gated[idx] = (bf16_t)(a / (1.0f + __expf(-g)));
  }
}

// ---------- launch ----------

extern "C" void kernel_launch(void* const* d_in, const int* in_sizes, int n_in,
                              void* d_out, int out_size, void* d_ws,
                              size_t ws_size, hipStream_t stream) {
  (void)in_sizes; (void)n_in; (void)out_size; (void)ws_size;
  const float* x   = (const float*)d_in[0];
  const float* Wq  = (const float*)d_in[1];
  const float* Wk  = (const float*)d_in[2];
  const float* Wv  = (const float*)d_in[3];
  const float* Wo  = (const float*)d_in[4];
  const float* qnw = (const float*)d_in[5];
  const float* knw = (const float*)d_in[6];

  const int M = BATCH_B * SEQ_T;        // 4096 rows
  char* ws = (char*)d_ws;
  size_t off = 0;
  auto carve = [&](size_t bytes) -> void* {
    void* p = ws + off;
    off += (bytes + 255) & ~(size_t)255;
    return p;
  };

  bf16_t* xb  = (bf16_t*)carve((size_t)M * EMBD * 2);
  bf16_t* wqb = (bf16_t*)carve((size_t)4096 * EMBD * 2);
  bf16_t* wkb = (bf16_t*)carve((size_t)512 * EMBD * 2);
  bf16_t* wvb = (bf16_t*)carve((size_t)512 * EMBD * 2);
  bf16_t* wob = (bf16_t*)carve((size_t)EMBD * EMBD * 2);
  float*  qgf = (float*)carve((size_t)M * 4096 * 4);
  float*  kf  = (float*)carve((size_t)M * 512 * 4);
  float*  vf  = (float*)carve((size_t)M * 512 * 4);
  bf16_t* qb  = (bf16_t*)carve((size_t)M * N_HEADS * HEAD_DIM * 2 / 1);
  bf16_t* kb  = (bf16_t*)carve((size_t)M * KV_HEADS * HEAD_DIM * 2);
  bf16_t* vb  = (bf16_t*)carve((size_t)M * KV_HEADS * HEAD_DIM * 2);
  float*  att = (float*)carve((size_t)M * EMBD * 4);
  bf16_t* gtd = (bf16_t*)carve((size_t)M * EMBD * 2);

  // 1) fp32 -> bf16
  {
    int n;
    n = M * EMBD;      ga_cvt_bf16<<<(n + 255) / 256, 256, 0, stream>>>(x, xb, n);
    n = 4096 * EMBD;   ga_cvt_bf16<<<(n + 255) / 256, 256, 0, stream>>>(Wq, wqb, n);
    n = 512 * EMBD;    ga_cvt_bf16<<<(n + 255) / 256, 256, 0, stream>>>(Wk, wkb, n);
    n = 512 * EMBD;    ga_cvt_bf16<<<(n + 255) / 256, 256, 0, stream>>>(Wv, wvb, n);
    n = EMBD * EMBD;   ga_cvt_bf16<<<(n + 255) / 256, 256, 0, stream>>>(Wo, wob, n);
  }

  // 2) projections (WMMA GEMMs)
  ga_gemm_wmma<<<dim3(M / 64, 4096 / 64), 128, 0, stream>>>(xb, wqb, qgf, M, 4096, EMBD);
  ga_gemm_wmma<<<dim3(M / 64, 512 / 64), 128, 0, stream>>>(xb, wkb, kf, M, 512, EMBD);
  ga_gemm_wmma<<<dim3(M / 64, 512 / 64), 128, 0, stream>>>(xb, wvb, vf, M, 512, EMBD);

  // 3) RMSNorm + RoPE + repack head-major bf16
  ga_norm_rope<<<dim3(M, N_HEADS + 2 * KV_HEADS), 128, 0, stream>>>(
      qgf, kf, vf, qnw, knw, qb, kb, vb);

  // 4) causal flash attention with WMMA
  ga_flash_attn<<<dim3(SEQ_T / 64, N_HEADS, BATCH_B), 128, 0, stream>>>(
      qb, kb, vb, att);

  // 5) sigmoid gate -> bf16
  {
    int n = M * EMBD;
    ga_gate<<<(n + 255) / 256, 256, 0, stream>>>(att, qgf, gtd, n);
  }

  // 6) output projection into d_out (f32)
  ga_gemm_wmma<<<dim3(M / 64, EMBD / 64), 128, 0, stream>>>(
      gtd, wob, (float*)d_out, M, EMBD, EMBD);
}
